// AttentionGNNModule_4269197492628
// MI455X (gfx1250) — compile-verified, matrix operand order checked by reading, not run on hardware
//
#include <hip/hip_runtime.h>
#include <stdint.h>

#define D      768
#define DOUT   384
#define LEN    512
#define BATCH  4
#define KK     51
#define BL     (BATCH * LEN)

#define MT 2   // 16-row tiles per wave (32 rows)
#define NT 4   // 16-col tiles per wave (64 cols)

typedef __attribute__((ext_vector_type(16))) __bf16 v16bf;
typedef __attribute__((ext_vector_type(8)))  float  v8f;

// ---------------------------------------------------------------- helpers

__device__ __forceinline__ unsigned short f2bf(float f) {
  unsigned u = __float_as_uint(f);
  u += 0x7FFFu + ((u >> 16) & 1u);   // round-to-nearest-even
  return (unsigned short)(u >> 16);
}

__device__ __forceinline__ float blockSum256(float v, float* red) {
  int wave = threadIdx.x >> 5, lane = threadIdx.x & 31;
#pragma unroll
  for (int o = 16; o > 0; o >>= 1) v += __shfl_xor(v, o, 32);
  if (lane == 0) red[wave] = v;
  __syncthreads();
  float t = (lane < 8) ? red[lane] : 0.f;
  if (wave == 0) {
#pragma unroll
    for (int o = 4; o > 0; o >>= 1) t += __shfl_xor(t, o, 32);
    if (lane == 0) red[0] = t;
  }
  __syncthreads();
  float r = red[0];
  __syncthreads();
  return r;
}

__device__ __forceinline__ float blockMax256(float v, float* red) {
  int wave = threadIdx.x >> 5, lane = threadIdx.x & 31;
#pragma unroll
  for (int o = 16; o > 0; o >>= 1) v = fmaxf(v, __shfl_xor(v, o, 32));
  if (lane == 0) red[wave] = v;
  __syncthreads();
  float t = (lane < 8) ? red[lane] : -3.402823466e38f;
  if (wave == 0) {
#pragma unroll
    for (int o = 4; o > 0; o >>= 1) t = fmaxf(t, __shfl_xor(t, o, 32));
    if (lane == 0) red[0] = t;
  }
  __syncthreads();
  float r = red[0];
  __syncthreads();
  return r;
}

// ------------------------------------------------- prep: f32 -> bf16 copies

__global__ void cvt_bf16_kernel(const float* __restrict__ in,
                                unsigned short* __restrict__ out, int n) {
  int i = blockIdx.x * blockDim.x + threadIdx.x;
  if (i < n) out[i] = f2bf(in[i]);
}

// W[k][n] (f32, DxD) -> Wt[n][k] (bf16, DxD)
__global__ void transpose_cvt_kernel(const float* __restrict__ W,
                                     unsigned short* __restrict__ Wt) {
  __shared__ float tile[32][33];
  int k0 = blockIdx.y * 32, n0 = blockIdx.x * 32;
  for (int r = threadIdx.y; r < 32; r += 8)
    tile[r][threadIdx.x] = W[(size_t)(k0 + r) * D + (n0 + threadIdx.x)];
  __syncthreads();
  for (int r = threadIdx.y; r < 32; r += 8)
    Wt[(size_t)(n0 + r) * D + (k0 + threadIdx.x)] = f2bf(tile[threadIdx.x][r]);
}

// ------------------------------------------------- WMMA GEMM: C = A*B + bias
// A: [M][Kd] bf16 row-major.  Bt: [N][Kd] bf16 (weight transposed so each
// lane's B operand chunks are contiguous 16B).  One wave computes a 32x64
// output strip (2 M-tiles x 4 N-tiles): A fragments reused across 4 WMMAs,
// B fragments across 2 -> 1.5 b128 loads per WMMA instead of 4.
// Per-lane operand layout (both A and B, 16x16x32 bf16):
//   base = (lane>=16) ? 8 : 0 ;  halves 0..7  -> K=base..base+7,
//                                halves 8..15 -> K=base+16..base+23.
__global__ void __launch_bounds__(256) gemm_bf16_wmma_kernel(
    const unsigned short* __restrict__ A,
    const unsigned short* __restrict__ Bt,
    const float* __restrict__ bias,
    float* __restrict__ C, int M, int N, int Kd) {
  int wid  = (blockIdx.x * blockDim.x + threadIdx.x) >> 5;
  int lane = threadIdx.x & 31;
  int stripsN = N >> 6;                      // 64-wide N strips
  int tm = wid / stripsN;                    // 32-row M strip
  int tn = wid - tm * stripsN;
  if (tm * 32 >= M) return;                  // wave-uniform guard
  int r16 = lane & 15;
  int hi  = lane >> 4;

  const unsigned short* arow[MT];
#pragma unroll
  for (int mi = 0; mi < MT; ++mi)
    arow[mi] = A + (size_t)(tm * 32 + mi * 16 + r16) * Kd + hi * 8;
  const unsigned short* brow[NT];
#pragma unroll
  for (int ni = 0; ni < NT; ++ni)
    brow[ni] = Bt + (size_t)(tn * 64 + ni * 16 + r16) * Kd + hi * 8;

  const v8f vzero = {0.f, 0.f, 0.f, 0.f, 0.f, 0.f, 0.f, 0.f};
  v8f acc[MT][NT];
#pragma unroll
  for (int mi = 0; mi < MT; ++mi)
#pragma unroll
    for (int ni = 0; ni < NT; ++ni) acc[mi][ni] = vzero;

  for (int k0 = 0; k0 < Kd; k0 += 32) {
    union { uint4 q[2]; v16bf v; } av[MT], bv[NT];
#pragma unroll
    for (int mi = 0; mi < MT; ++mi) {
      av[mi].q[0] = *(const uint4*)(arow[mi] + k0);
      av[mi].q[1] = *(const uint4*)(arow[mi] + k0 + 16);
    }
#pragma unroll
    for (int ni = 0; ni < NT; ++ni) {
      bv[ni].q[0] = *(const uint4*)(brow[ni] + k0);
      bv[ni].q[1] = *(const uint4*)(brow[ni] + k0 + 16);
    }
#pragma unroll
    for (int mi = 0; mi < MT; ++mi)
#pragma unroll
      for (int ni = 0; ni < NT; ++ni)
        acc[mi][ni] = __builtin_amdgcn_wmma_f32_16x16x32_bf16(
            /*neg_a=*/false, av[mi].v, /*neg_b=*/false, bv[ni].v,
            /*c_mod=*/(short)0, acc[mi][ni],
            /*reuse_a=*/false, /*reuse_b=*/false);
  }

#pragma unroll
  for (int mi = 0; mi < MT; ++mi) {
#pragma unroll
    for (int ni = 0; ni < NT; ++ni) {
      int col = tn * 64 + ni * 16 + r16;
      float bb = bias[col];
#pragma unroll
      for (int r = 0; r < 8; ++r) {
        int row = tm * 32 + mi * 16 + hi * 8 + r;   // VGPR r: M = r + 8*hi
        C[(size_t)row * N + col] = acc[mi][ni][r] + bb;
      }
    }
  }
}

// ------------------------------------------------- top-51 of each 512-row
__global__ void __launch_bounds__(32) topk_kernel(
    const float* __restrict__ attn, int* __restrict__ idx_out,
    float* __restrict__ val_out) {
  int row  = blockIdx.x;                   // 0..BL-1
  int lane = threadIdx.x;
  const float* a = attn + (size_t)row * LEN;
  float v[16];
#pragma unroll
  for (int t = 0; t < 16; ++t) v[t] = a[t * 32 + lane];
  for (int i = 0; i < KK; ++i) {
    float bv = v[0]; int bt = 0;
#pragma unroll
    for (int t = 1; t < 16; ++t)
      if (v[t] > bv) { bv = v[t]; bt = t; }
    int bidx = bt * 32 + lane;
#pragma unroll
    for (int o = 16; o > 0; o >>= 1) {     // argmax butterfly, tie -> low idx
      float ov = __shfl_xor(bv, o, 32);
      int   oi = __shfl_xor(bidx, o, 32);
      if (ov > bv || (ov == bv && oi < bidx)) { bv = ov; bidx = oi; }
    }
    if (lane == 0) { idx_out[row * KK + i] = bidx; val_out[row * KK + i] = bv; }
    int kt = bidx >> 5, kl = bidx & 31;
    if (lane == kl) {
#pragma unroll
      for (int t = 0; t < 16; ++t)
        if (t == kt) v[t] = -3.402823466e38f;
    }
  }
}

// ------------------------------------------------- GAT attention + gate
__global__ void __launch_bounds__(256) gat_kernel(
    const float* __restrict__ feat_src, const float* __restrict__ feat_dst,
    const float* __restrict__ hidden, const float* __restrict__ attn_vec,
    const float* __restrict__ gat_bias, const int* __restrict__ tk_idx,
    const float* __restrict__ tk_val, const float* __restrict__ gate_W,
    const float* __restrict__ gate_b, float* __restrict__ h_out,
    float* __restrict__ gate_out) {
  int bl = blockIdx.x;                       // 0..BL-1
  int b  = bl >> 9;                          // /LEN
  int tid = threadIdx.x;
  int wave = tid >> 5, lane = tid & 31;
  __shared__ float fdst[D];
  __shared__ float sav[D];
  __shared__ float sc[64];
  __shared__ float red[8];
  const float* fd = feat_dst + (size_t)bl * D;
  for (int d = tid; d < D; d += 256) { fdst[d] = fd[d]; sav[d] = attn_vec[d]; }
  __syncthreads();

  // scores[k] = attn_vec . leaky_relu(g_k + fdst, 0.2), one wave per k
  for (int k = wave; k < KK; k += 8) {
    int j = tk_idx[bl * KK + k];
    const float* g = feat_src + (size_t)(b * LEN + j) * D;
    float s = 0.f;
    for (int d = lane; d < D; d += 32) {
      float e = g[d] + fdst[d];
      e = (e > 0.f) ? e : 0.2f * e;
      s += e * sav[d];
    }
#pragma unroll
    for (int o = 16; o > 0; o >>= 1) s += __shfl_xor(s, o, 32);
    if (lane == 0)
      sc[k] = (tk_val[bl * KK + k] > 0.f) ? s : -1e9f;
  }
  __syncthreads();

  // softmax over the 51 scores (wave 0)
  if (wave == 0) {
    float a0 = (lane < KK)      ? sc[lane]      : -1e30f;
    float a1 = (lane + 32 < KK) ? sc[lane + 32] : -1e30f;
    float m = fmaxf(a0, a1);
#pragma unroll
    for (int o = 16; o > 0; o >>= 1) m = fmaxf(m, __shfl_xor(m, o, 32));
    float e0 = (lane < KK)      ? __expf(a0 - m) : 0.f;
    float e1 = (lane + 32 < KK) ? __expf(a1 - m) : 0.f;
    float s = e0 + e1;
#pragma unroll
    for (int o = 16; o > 0; o >>= 1) s += __shfl_xor(s, o, 32);
    float inv = 1.f / s;
    if (lane < KK)      sc[lane]      = e0 * inv;
    if (lane + 32 < KK) sc[lane + 32] = e1 * inv;
  }
  __syncthreads();

  // h = sum_k a_k * g_k ; each thread owns d = tid, tid+256, tid+512
  float a0 = 0.f, a1 = 0.f, a2 = 0.f;
  for (int k = 0; k < KK; ++k) {
    int j = tk_idx[bl * KK + k];
    const float* g = feat_src + (size_t)(b * LEN + j) * D;
    if (k + 1 < KK) {
      int jn = tk_idx[bl * KK + k + 1];
      __builtin_prefetch(feat_src + (size_t)(b * LEN + jn) * D + tid, 0, 0);
    }
    float p = sc[k];
    a0 += p * g[tid];
    a1 += p * g[tid + 256];
    a2 += p * g[tid + 512];
  }
  const float* hs = hidden + (size_t)bl * D;
  float h0 = a0 + hs[tid]       + gat_bias[tid];
  float h1 = a1 + hs[tid + 256] + gat_bias[tid + 256];
  float h2 = a2 + hs[tid + 512] + gat_bias[tid + 512];
  h0 = (h0 > 0.f) ? h0 : 0.01f * h0;
  h1 = (h1 > 0.f) ? h1 : 0.01f * h1;
  h2 = (h2 > 0.f) ? h2 : 0.01f * h2;
  float* ho = h_out + (size_t)bl * D;
  ho[tid] = h0; ho[tid + 256] = h1; ho[tid + 512] = h2;

  // gate scalar = h . gate_W + gate_b
  float gp = h0 * gate_W[tid] + h1 * gate_W[tid + 256] + h2 * gate_W[tid + 512];
#pragma unroll
  for (int o = 16; o > 0; o >>= 1) gp += __shfl_xor(gp, o, 32);
  if (lane == 0) red[wave] = gp;
  __syncthreads();
  if (tid == 0) {
    float t = 0.f;
    for (int w = 0; w < 8; ++w) t += red[w];
    gate_out[bl] = t + gate_b[0];
  }
}

// ------------------------------------------------- pooling + LN + FC + LN
__global__ void __launch_bounds__(256) pool_kernel(
    const float* __restrict__ h, const float* __restrict__ gate,
    const float* __restrict__ ln_g, const float* __restrict__ ln_b,
    const float* __restrict__ fc_W, const float* __restrict__ fc_b,
    const float* __restrict__ ln2_g, const float* __restrict__ ln2_b,
    float* __restrict__ out) {
  int b = blockIdx.x;
  int tid = threadIdx.x;
  __shared__ float p[LEN];
  __shared__ float x[D];
  __shared__ float y[DOUT];
  __shared__ float red[8];

  // softmax over gate along L
  const float* gt = gate + b * LEN;
  float g0 = gt[tid], g1 = gt[tid + 256];
  float m = blockMax256(fmaxf(g0, g1), red);
  float e0 = __expf(g0 - m), e1 = __expf(g1 - m);
  float s = blockSum256(e0 + e1, red);
  float inv = 1.f / s;
  p[tid] = e0 * inv; p[tid + 256] = e1 * inv;
  __syncthreads();

  // pooled = sum_l p_l * h_l  (each thread owns 3 dims)
  float a0 = 0.f, a1 = 0.f, a2 = 0.f;
  const float* hb = h + (size_t)b * LEN * D;
  for (int l = 0; l < LEN; ++l) {
    float pl = p[l];
    const float* r = hb + (size_t)l * D;
    a0 += pl * r[tid]; a1 += pl * r[tid + 256]; a2 += pl * r[tid + 512];
  }
  a0 = fmaxf(a0, 0.f); a1 = fmaxf(a1, 0.f); a2 = fmaxf(a2, 0.f);   // relu

  // layer norm over D
  float mu = blockSum256(a0 + a1 + a2, red) * (1.f / (float)D);
  float d0 = a0 - mu, d1 = a1 - mu, d2 = a2 - mu;
  float var = blockSum256(d0 * d0 + d1 * d1 + d2 * d2, red) * (1.f / (float)D);
  float rstd = rsqrtf(var + 1e-5f);
  x[tid]       = d0 * rstd * ln_g[tid]       + ln_b[tid];
  x[tid + 256] = d1 * rstd * ln_g[tid + 256] + ln_b[tid + 256];
  x[tid + 512] = d2 * rstd * ln_g[tid + 512] + ln_b[tid + 512];
  __syncthreads();

  // FC: (D) @ (D x DOUT)
  for (int n = tid; n < DOUT; n += 256) {
    float s2 = 0.f;
    for (int d = 0; d < D; ++d) s2 += x[d] * fc_W[(size_t)d * DOUT + n];
    y[n] = s2 + fc_b[n];
  }
  __syncthreads();

  // layer norm over DOUT
  float v0 = (tid < DOUT) ? y[tid] : 0.f;
  float v1 = (tid + 256 < DOUT) ? y[tid + 256] : 0.f;
  float mu2 = blockSum256(v0 + v1, red) * (1.f / (float)DOUT);
  float q0 = (tid < DOUT) ? (v0 - mu2) * (v0 - mu2) : 0.f;
  float q1 = (tid + 256 < DOUT) ? (v1 - mu2) * (v1 - mu2) : 0.f;
  float var2 = blockSum256(q0 + q1, red) * (1.f / (float)DOUT);
  float rstd2 = rsqrtf(var2 + 1e-5f);
  for (int n = tid; n < DOUT; n += 256)
    out[b * DOUT + n] = (y[n] - mu2) * rstd2 * ln2_g[n] + ln2_b[n];
}

// ------------------------------------------------- launch

extern "C" void kernel_launch(void* const* d_in, const int* in_sizes, int n_in,
                              void* d_out, int out_size, void* d_ws,
                              size_t ws_size, hipStream_t stream) {
  const float* hidden    = (const float*)d_in[0];
  const float* attention = (const float*)d_in[1];
  const float* W_src     = (const float*)d_in[2];
  const float* b_src     = (const float*)d_in[3];
  const float* W_dst     = (const float*)d_in[4];
  const float* b_dst     = (const float*)d_in[5];
  const float* attn_vec  = (const float*)d_in[6];
  const float* gat_bias  = (const float*)d_in[7];
  const float* gate_W    = (const float*)d_in[8];
  const float* gate_b    = (const float*)d_in[9];
  const float* ln_g      = (const float*)d_in[10];
  const float* ln_b      = (const float*)d_in[11];
  const float* fc_W      = (const float*)d_in[12];
  const float* fc_b      = (const float*)d_in[13];
  const float* ln2_g     = (const float*)d_in[14];
  const float* ln2_b     = (const float*)d_in[15];
  float* out = (float*)d_out;

  char* ws = (char*)d_ws;
  size_t off = 0;
  auto carve = [&](size_t bytes) -> void* {
    void* ptr = ws + off;
    off += (bytes + 255) & ~(size_t)255;
    return ptr;
  };
  unsigned short* hid_bf   = (unsigned short*)carve((size_t)BL * D * 2);
  unsigned short* wsrc_t   = (unsigned short*)carve((size_t)D * D * 2);
  unsigned short* wdst_t   = (unsigned short*)carve((size_t)D * D * 2);
  float*          feat_src = (float*)carve((size_t)BL * D * 4);
  float*          feat_dst = (float*)carve((size_t)BL * D * 4);
  int*            tk_idx   = (int*)carve((size_t)BL * KK * 4);
  float*          tk_val   = (float*)carve((size_t)BL * KK * 4);
  float*          h_buf    = (float*)carve((size_t)BL * D * 4);
  float*          gate_buf = (float*)carve((size_t)BL * 4);

  // 1) bf16 conversions / weight transposes
  cvt_bf16_kernel<<<(BL * D + 255) / 256, 256, 0, stream>>>(hidden, hid_bf,
                                                            BL * D);
  dim3 tb(32, 8);
  transpose_cvt_kernel<<<dim3(D / 32, D / 32), tb, 0, stream>>>(W_src, wsrc_t);
  transpose_cvt_kernel<<<dim3(D / 32, D / 32), tb, 0, stream>>>(W_dst, wdst_t);

  // 2) WMMA GEMMs: feat_src / feat_dst (32x64 strip per wave)
  int totalWaves = (BL / 32) * (D / 64);           // 768
  int gemmBlocks = (totalWaves * 32) / 256;        // 96 blocks of 8 waves
  gemm_bf16_wmma_kernel<<<gemmBlocks, 256, 0, stream>>>(
      hid_bf, wsrc_t, b_src, feat_src, BL, D, D);
  gemm_bf16_wmma_kernel<<<gemmBlocks, 256, 0, stream>>>(
      hid_bf, wdst_t, b_dst, feat_dst, BL, D, D);

  // 3) top-K
  topk_kernel<<<BL, 32, 0, stream>>>(attention, tk_idx, tk_val);

  // 4) GAT attention + gate
  gat_kernel<<<BL, 256, 0, stream>>>(feat_src, feat_dst, hidden, attn_vec,
                                     gat_bias, tk_idx, tk_val, gate_W, gate_b,
                                     h_buf, gate_buf);

  // 5) pooling + LN + FC + LN
  pool_kernel<<<BATCH, 256, 0, stream>>>(h_buf, gate_buf, ln_g, ln_b, fc_W,
                                         fc_b, ln2_g, ln2_b, out);
}
